// Conv2d_XnorPP_87419764342823
// MI455X (gfx1250) — 8-bit matrix operand order repaired and checked by reading, compiled, not run on hardware
//
#include <hip/hip_runtime.h>


typedef int   v16i __attribute__((ext_vector_type(16)));
typedef float v8f  __attribute__((ext_vector_type(8)));

#define B_    32
#define CI_   256
#define CO_   256
#define H_    56
#define HP_   58
#define WP_   66
#define KTAPS 9

#define XP_BYTES ((size_t)B_ * HP_ * WP_ * CI_)
#define WB_BYTES ((size_t)CO_ * KTAPS * CI_)

#define FP8_P1 0x38u
#define FP8_M1 0xB8u

#define KC_CHUNK 128
#define XS_PITCH 144
#define AS_PITCH 144
#define XS_BYTES (3 * WP_ * XS_PITCH)
#define AS_BYTES (128 * AS_PITCH)

__global__ __launch_bounds__(256) void zero_bytes16(int4* __restrict__ p, int n16) {
  int i = blockIdx.x * 256 + threadIdx.x;
  if (i < n16) {
    int4 z; z.x = 0; z.y = 0; z.z = 0; z.w = 0;
    p[i] = z;
  }
}

__global__ __launch_bounds__(256) void binarize_x(const float* __restrict__ x,
                                                  signed char* __restrict__ Xp) {
  __shared__ __align__(16) signed char L[H_ * 272];
  const int tid = threadIdx.x;
  const int bh  = blockIdx.x;
  const int b = bh / H_, h = bh % H_;
  for (int i = tid; i < CI_ * H_; i += 256) {
    const int c = i / H_, w = i % H_;
    const float v = x[(((size_t)b * CI_ + c) * H_ + h) * H_ + w];
    L[w * 272 + c] = (signed char)((v > 0.0f) ? FP8_P1 : FP8_M1);
  }
  __syncthreads();
  const size_t obase = (((size_t)b * HP_ + h + 1) * WP_ + 1) * CI_;
  for (int i = tid; i < H_ * (CI_ / 16); i += 256) {
    const int w = i >> 4, cg = i & 15;
    const int4 v = *(const int4*)(L + w * 272 + cg * 16);
    *(int4*)(Xp + obase + (size_t)w * CI_ + cg * 16) = v;
  }
}

__global__ __launch_bounds__(256) void binarize_w(const float* __restrict__ wgt,
                                                  signed char* __restrict__ Wb) {
  const int i = blockIdx.x * 256 + threadIdx.x;
  if (i >= CO_ * KTAPS * (CI_ / 4)) return;
  const int ci4 = i & 63;
  const int tap = (i >> 6) % KTAPS;
  const int co  = i / (64 * KTAPS);
  const int kh = tap / 3, kw = tap % 3;
  unsigned int packed = 0;
  #pragma unroll
  for (int j = 0; j < 4; ++j) {
    const float v = wgt[(((size_t)co * CI_ + ci4 * 4 + j) * 3 + kh) * 3 + kw];
    packed |= ((v > 0.0f) ? FP8_P1 : FP8_M1) << (8 * j);
  }
  *(unsigned int*)(Wb + (size_t)co * (KTAPS * CI_) + tap * CI_ + ci4 * 4) = packed;
}

__global__ __launch_bounds__(256) void xnor_gemm_fp8(
    const signed char* __restrict__ Xp,
    const signed char* __restrict__ Wb,
    const float* __restrict__ alpha,
    const float* __restrict__ beta,
    const float* __restrict__ gamma,
    float* __restrict__ out) {
  __shared__ __align__(16) signed char Xs[XS_BYTES];
  __shared__ __align__(16) signed char As[AS_BYTES];

  const int tid    = threadIdx.x;
  const int lane   = tid & 31;
  const int wave   = tid >> 5;
  const int laneN  = lane & 15;
  const int laneHi = lane >> 4;
  const int bh = blockIdx.x;
  const int b = bh / H_, h = bh % H_;
  const int co_block = blockIdx.y * 128;

  v8f acc[4] = {};

  for (int kc = 0; kc < 2; ++kc) {
    __syncthreads();
    {
      const size_t gbase = ((size_t)(b * HP_ + h) * WP_) * CI_ + (size_t)kc * KC_CHUNK;
      for (int i = tid; i < (3 * WP_ * KC_CHUNK) / 16; i += 256) {
        const int ci16 = (i & 7) * 16;
        const int rw   = i >> 3;
        const int4 v = *(const int4*)(Xp + gbase + (size_t)rw * CI_ + ci16);
        *(int4*)(Xs + rw * XS_PITCH + ci16) = v;
      }
    }
    for (int tap = 0; tap < KTAPS; ++tap) {
      __syncthreads();
      {
        const size_t wbase = (size_t)co_block * (KTAPS * CI_) + tap * CI_ + kc * KC_CHUNK;
        for (int i = tid; i < (128 * KC_CHUNK) / 16; i += 256) {
          const int k16 = (i & 7) * 16;
          const int col = i >> 3;
          const int4 v = *(const int4*)(Wb + wbase + (size_t)col * (KTAPS * CI_) + k16);
          *(int4*)(As + col * AS_PITCH + k16) = v;
        }
      }
      __syncthreads();
      const int kh = tap / 3, kw = tap % 3;

      v16i a;
      {
        const signed char* ap = As + (wave * 16 + laneN) * AS_PITCH + laneHi * 16;
        #pragma unroll
        for (int hk = 0; hk < 2; ++hk) {
          const signed char* q = ap + hk * 64;
          const int2 p0 = *(const int2*)(q + 0);
          const int2 p1 = *(const int2*)(q + 8);
          const int2 p2 = *(const int2*)(q + 32);
          const int2 p3 = *(const int2*)(q + 40);
          a[hk * 8 + 0] = p0.x; a[hk * 8 + 1] = p0.y;
          a[hk * 8 + 2] = p1.x; a[hk * 8 + 3] = p1.y;
          a[hk * 8 + 4] = p2.x; a[hk * 8 + 5] = p2.y;
          a[hk * 8 + 6] = p3.x; a[hk * 8 + 7] = p3.y;
        }
      }

      v16i bf[4];
      #pragma unroll
      for (int t = 0; t < 4; ++t) {
        const int wrow = t * 16 + laneN + kw;
        const signed char* bp = Xs + (kh * WP_ + wrow) * XS_PITCH + laneHi * 16;
        #pragma unroll
        for (int g = 0; g < 4; ++g) {
          const int4 v = *(const int4*)(bp + g * 32);
          bf[t][g * 4 + 0] = v.x; bf[t][g * 4 + 1] = v.y;
          bf[t][g * 4 + 2] = v.z; bf[t][g * 4 + 3] = v.w;
        }
      }
      #pragma unroll
      for (int t = 0; t < 4; ++t) {
        acc[t] = __builtin_amdgcn_wmma_f32_16x16x128_fp8_fp8(
            a, bf[t], (short)0, acc[t], false, false);
      }
    }
  }

  const float bscale = beta[h];
  #pragma unroll
  for (int t = 0; t < 4; ++t) {
    const int w = t * 16 + laneN;
    if (w < H_) {
      const float g = gamma[w] * bscale;
      #pragma unroll
      for (int r = 0; r < 8; ++r) {
        const int co = co_block + wave * 16 + laneHi * 8 + r;
        out[(((size_t)b * CO_ + co) * H_ + h) * H_ + w] = acc[t][r] * alpha[co] * g;
      }
    }
  }
}

extern "C" void kernel_launch(void* const* d_in, const int* in_sizes, int n_in,
                              void* d_out, int out_size, void* d_ws, size_t ws_size,
                              hipStream_t stream) {
  const float* x     = (const float*)d_in[0];
  const float* wgt   = (const float*)d_in[1];
  const float* alpha = (const float*)d_in[2];
  const float* beta  = (const float*)d_in[3];
  const float* gamma = (const float*)d_in[4];
  float* out = (float*)d_out;

  signed char* Xp = (signed char*)d_ws;
  signed char* Wb = Xp + XP_BYTES;

  const int n16 = (int)(XP_BYTES / 16);
  zero_bytes16<<<(n16 + 255) / 256, 256, 0, stream>>>((int4*)Xp, n16);

  binarize_x<<<B_ * H_, 256, 0, stream>>>(x, Xp);

  const int nw = CO_ * KTAPS * (CI_ / 4);
  binarize_w<<<(nw + 255) / 256, 256, 0, stream>>>(wgt, Wb);

  dim3 grid(B_ * H_, CO_ / 128);
  xnor_gemm_fp8<<<grid, 256, 0, stream>>>(Xp, Wb, alpha, beta, gamma, out);
}
